// LaSAForNERV2_14310831030955
// MI455X (gfx1250) — matrix operand order repaired and checked by reading, not compiled, not run on hardware
//
#include <hip/hip_runtime.h>
#include <math.h>

typedef __bf16 bf16;
typedef __attribute__((ext_vector_type(8)))  __bf16 v8bf;
typedef __attribute__((ext_vector_type(16))) __bf16 v16bf;
typedef __attribute__((ext_vector_type(8)))  float  v8f;
typedef __attribute__((ext_vector_type(4)))  int    v4i;

#define B_    8
#define L_    512
#define H_    768
#define NL_   4
#define NH_   12
#define FF_   3072
#define MCN_  128
#define MCL_  4
#define NLAB_ 9
#define CHNH_ 8
#define TOK_  (B_*L_)   // 4096

#ifndef __has_builtin
#define __has_builtin(x) 0
#endif
#if __has_builtin(__builtin_amdgcn_global_load_async_to_lds_b128) && \
    __has_builtin(__builtin_amdgcn_s_wait_asynccnt)
#define USE_ASYNC 1
#else
#define USE_ASYNC 0
#endif

#define AS1 __attribute__((address_space(1)))
#define AS3 __attribute__((address_space(3)))

#define DEVFN static __device__ __forceinline__

DEVFN float wsum(float v){ for (int o=16;o;o>>=1) v += __shfl_xor(v,o,32); return v; }
DEVFN float wmax(float v){ for (int o=16;o;o>>=1) v = fmaxf(v,__shfl_xor(v,o,32)); return v; }

// A-fragment (16x32 bf16): lane l = row (l&15); K-base = 16*(l>>4).
// elements 0..15 -> K = kb+0..15 (contiguous, matches B)
DEVFN v16bf load_afrag(const bf16* p){
  v8bf lo = *(const v8bf*)p;
  v8bf hi = *(const v8bf*)(p + 8);
  return __builtin_shufflevector(lo,hi,0,1,2,3,4,5,6,7,8,9,10,11,12,13,14,15);
}
#define WMMA_BF16(a,b,c) \
  __builtin_amdgcn_wmma_f32_16x16x32_bf16(false,(a),false,(b),(short)0,(c),false,false)

// ---------------------------------------------------------------- weight prep
__global__ void k_cast_transpose(const float* __restrict__ w, bf16* __restrict__ o,
                                 int K, int N){
  __shared__ float t[32][33];
  int kb = blockIdx.x*32, nb = blockIdx.y*32;
  int tx = threadIdx.x & 31, ty = threadIdx.x >> 5;      // 256 thr -> ty 0..7
  for (int i = ty; i < 32; i += 8){
    int k = kb+i, n = nb+tx;
    t[i][tx] = (k < K && n < N) ? w[(size_t)k*N + n] : 0.f;
  }
  __syncthreads();
  for (int i = ty; i < 32; i += 8){
    int n = nb+i, k = kb+tx;
    if (n < N && k < K) o[(size_t)n*K + k] = (bf16)t[tx][i];
  }
}

// ---------------------------------------------------------------- embeddings
__global__ void k_embed(const int* __restrict__ ids, const int* __restrict__ tt,
                        const float* __restrict__ we, const float* __restrict__ pe,
                        const float* __restrict__ te, float* __restrict__ out,
                        int Lseq, int H){
  int t = blockIdx.x;
  int l = t % Lseq;
  int id = ids[t], ty = tt[t];
  for (int h = threadIdx.x; h < H; h += blockDim.x)
    out[(size_t)t*H + h] = we[(size_t)id*H + h] + pe[(size_t)l*H + h] + te[(size_t)ty*H + h];
}

// ---------------------------------------------------------------- layernorm
__global__ void k_ln(const float* __restrict__ in, const bf16* __restrict__ res,
                     const float* __restrict__ g, const float* __restrict__ b,
                     bf16* __restrict__ out, int H){
  __shared__ float rs[8], rs2[8];
  int row = blockIdx.x;
  const float* x = in + (size_t)row*H;
  const bf16*  r = res ? res + (size_t)row*H : nullptr;
  float s = 0.f, s2 = 0.f;
  for (int h = threadIdx.x; h < H; h += blockDim.x){
    float v = x[h] + (r ? (float)r[h] : 0.f);
    s += v; s2 += v*v;
  }
  s = wsum(s); s2 = wsum(s2);
  int wid = threadIdx.x>>5, lane = threadIdx.x&31;
  if (lane == 0){ rs[wid]=s; rs2[wid]=s2; }
  __syncthreads();
  if (threadIdx.x == 0){
    float a=0.f, a2=0.f;
    for (int i=0;i<8;i++){ a+=rs[i]; a2+=rs2[i]; }
    rs[0]=a; rs2[0]=a2;
  }
  __syncthreads();
  float mean = rs[0]/H;
  float var  = rs2[0]/H - mean*mean;
  float rinv = rsqrtf(var + 1e-12f);
  for (int h = threadIdx.x; h < H; h += blockDim.x){
    float v = x[h] + (r ? (float)r[h] : 0.f);
    out[(size_t)row*H + h] = (bf16)((v-mean)*rinv*g[h] + b[h]);
  }
}

// ---------------------------------------------------------------- WMMA GEMM
// C[M,N] = A[M,K](bf16 rowmajor) @ WT[N,K](bf16) + bias ; epi 0=none 1=gelu
// Block = 8 waves: 128 rows x 64 cols. B panel (64 rows x 64 K per step) is
// staged in LDS and shared by all 8 waves (8x cut in B traffic), double-
// buffered via the CDNA5 async global->LDS path (ASYNCcnt) when available.
// One barrier covers 8 WMMAs (2 K-substeps x 4 N-tiles). Requires K % 64 == 0.
template<int EPI, int OUTB>
__global__ __launch_bounds__(256)
void k_gemm(const bf16* __restrict__ A, const bf16* __restrict__ WT,
            const float* __restrict__ bias, void* __restrict__ Cout,
            int M, int N, int K){
  __shared__ bf16 Bs[2][64*64];
  int wv = threadIdx.x >> 5, lane = threadIdx.x & 31;
  int cg = N >> 6;
  int bm = blockIdx.x / cg, bn = blockIdx.x % cg;
  int row0 = bm*128 + wv*16, col0 = bn << 6;
  int r = lane & 15, hi = lane >> 4;
  const bf16* ap = A + (size_t)(row0 + r)*K + hi*16;
  // per-thread 32B (two b128 chunks) of the 64x64 B panel:
  // row n = tid>>2 (0..63), K-base = (tid&3)*16
  int tn_ = threadIdx.x >> 2, tk = (threadIdx.x & 3) * 16;
  const bf16* bsrc = WT + (size_t)(col0 + tn_)*K + tk;
  int boff = tn_*64 + tk;                   // element offset inside Bs[cur]
  v8f acc[4] = {v8f{},v8f{},v8f{},v8f{}};
  int cur = 0;
#if USE_ASYNC
  __builtin_amdgcn_global_load_async_to_lds_b128(
      (AS1 v4i*)(bsrc),     (AS3 v4i*)(&Bs[0][boff]),     0, 0);
  __builtin_amdgcn_global_load_async_to_lds_b128(
      (AS1 v4i*)(bsrc + 8), (AS3 v4i*)(&Bs[0][boff + 8]), 0, 0);
  for (int k0 = 0; k0 < K; k0 += 64){
    __builtin_amdgcn_s_wait_asynccnt(0);
    __syncthreads();
    if (k0 + 64 < K){
      __builtin_amdgcn_global_load_async_to_lds_b128(
          (AS1 v4i*)(bsrc + k0 + 64),     (AS3 v4i*)(&Bs[cur^1][boff]),     0, 0);
      __builtin_amdgcn_global_load_async_to_lds_b128(
          (AS1 v4i*)(bsrc + k0 + 64 + 8), (AS3 v4i*)(&Bs[cur^1][boff + 8]), 0, 0);
    }
    __builtin_prefetch(ap + k0 + 256, 0, 0);
    const bf16* bb = &Bs[cur][0];
    for (int s = 0; s < 2; ++s){
      v16bf a = load_afrag(ap + k0 + s*32);
      for (int j = 0; j < 4; ++j){
        v16bf bj = *(const v16bf*)(bb + (j*16 + r)*64 + s*32 + hi*16);
        acc[j] = WMMA_BF16(a, bj, acc[j]);
      }
    }
    cur ^= 1;
  }
#else
  uint4 stg0 = *(const uint4*)bsrc;
  uint4 stg1 = *(const uint4*)(bsrc + 8);
  for (int k0 = 0; k0 < K; k0 += 64){
    *(uint4*)(&Bs[cur][boff])     = stg0;
    *(uint4*)(&Bs[cur][boff + 8]) = stg1;
    if (k0 + 64 < K){
      stg0 = *(const uint4*)(bsrc + k0 + 64);
      stg1 = *(const uint4*)(bsrc + k0 + 64 + 8);
    }
    __builtin_prefetch(ap + k0 + 256, 0, 0);
    __syncthreads();
    const bf16* bb = &Bs[cur][0];
    for (int s = 0; s < 2; ++s){
      v16bf a = load_afrag(ap + k0 + s*32);
      for (int j = 0; j < 4; ++j){
        v16bf bj = *(const v16bf*)(bb + (j*16 + r)*64 + s*32 + hi*16);
        acc[j] = WMMA_BF16(a, bj, acc[j]);
      }
    }
    cur ^= 1;
  }
#endif
  int orow = row0 + hi*8;
  for (int j = 0; j < 4; ++j){
    int col = col0 + j*16 + r;
    float bb = bias ? bias[col] : 0.f;
    for (int e = 0; e < 8; ++e){
      float v = acc[j][e] + bb;
      if (EPI == 1) v = 0.5f * v * (1.f + erff(v * 0.70710678118654752f));
      size_t idx = (size_t)(orow + e)*N + col;
      if (OUTB) ((bf16*)Cout)[idx] = (bf16)v;
      else      ((float*)Cout)[idx] = v;
    }
  }
}

// ---------------------------------------------------------------- QKV pack
__global__ void k_qkv_pack(const bf16* __restrict__ q, const bf16* __restrict__ k,
                           const bf16* __restrict__ v, bf16* __restrict__ qh,
                           bf16* __restrict__ kh, bf16* __restrict__ vT,
                           int S, int heads, int dh){
  size_t i = (size_t)blockIdx.x * blockDim.x + threadIdx.x;
  int d = i % dh; size_t t2 = i / dh;
  int h = t2 % heads; size_t t3 = t2 / heads;
  int s = t3 % S; int b = (int)(t3 / S);
  size_t bh = (size_t)b*heads + h;
  size_t o1 = (bh*S + s)*dh + d;
  qh[o1] = q[i];
  kh[o1] = k[i];
  vT[(bh*dh + d)*S + s] = v[i];
}

__global__ void k_ctx_unpack(const bf16* __restrict__ ctxh, bf16* __restrict__ ctx,
                             int S, int heads, int dh){
  size_t i = (size_t)blockIdx.x * blockDim.x + threadIdx.x;
  int d = i % dh; size_t t2 = i / dh;
  int h = t2 % heads; size_t t3 = t2 / heads;
  int s = t3 % S; int b = (int)(t3 / S);
  ctx[i] = ctxh[(((size_t)b*heads + h)*S + s)*dh + d];
}

// ------------------------------------------------- attention scores (WMMA)
// probs[bh,S,S] = (Q Kt^T)*scale + mask ; Q,Kt: [bh,S,dh] bf16
__global__ __launch_bounds__(256)
void k_attn_scores(const bf16* __restrict__ qh, const bf16* __restrict__ kh,
                   const float* __restrict__ maskB, bf16* __restrict__ probs,
                   int S, int dh, int heads, float scale){
  int bh = blockIdx.y; int b = bh / heads;
  const bf16* Q  = qh + (size_t)bh*S*dh;
  const bf16* Kt = kh + (size_t)bh*S*dh;
  bf16* P = probs + (size_t)bh*S*S;
  int wid  = blockIdx.x * (blockDim.x >> 5) + (threadIdx.x >> 5);
  int lane = threadIdx.x & 31;
  int cg = S >> 6;
  int tm = wid / cg, tn = wid % cg;
  int row0 = tm << 4, col0 = tn << 6;
  int r = lane & 15, hi = lane >> 4;
  const bf16* ap = Q  + (size_t)(row0 + r)*dh + hi*16;
  const bf16* bp = Kt + (size_t)(col0 + r)*dh + hi*16;
  v8f acc[4] = {v8f{},v8f{},v8f{},v8f{}};
  for (int k0 = 0; k0 < dh; k0 += 32){
    v16bf a = load_afrag(ap + k0);
    for (int j = 0; j < 4; ++j){
      v16bf bb = *(const v16bf*)(bp + (size_t)j*16*dh + k0);
      acc[j] = WMMA_BF16(a, bb, acc[j]);
    }
  }
  const float* mrow = maskB + (size_t)b*S;
  int orow = row0 + hi*8;
  for (int j = 0; j < 4; ++j){
    int col = col0 + j*16 + r;
    float madd = mrow[col];
    for (int e = 0; e < 8; ++e)
      P[(size_t)(orow + e)*S + col] = (bf16)(acc[j][e]*scale + madd);
  }
}

__global__ void k_softmax(bf16* __restrict__ probs, int S){
  int row  = blockIdx.x * (blockDim.x >> 5) + (threadIdx.x >> 5);
  int lane = threadIdx.x & 31;
  bf16* p = probs + (size_t)row*S;
  float mx = -1e30f;
  for (int i = lane; i < S; i += 32) mx = fmaxf(mx, (float)p[i]);
  mx = wmax(mx);
  float sum = 0.f;
  for (int i = lane; i < S; i += 32) sum += __expf((float)p[i]-mx);
  sum = wsum(sum);
  float inv = 1.f/sum;
  for (int i = lane; i < S; i += 32) p[i] = (bf16)(__expf((float)p[i]-mx)*inv);
}

// ctxh[bh,S,dh] = probs[bh,S,S] @ V ; vT: [bh,dh,S]
__global__ __launch_bounds__(256)
void k_attn_ctx(const bf16* __restrict__ probs, const bf16* __restrict__ vT,
                bf16* __restrict__ ctxh, int S, int dh){
  int bh = blockIdx.y;
  const bf16* A  = probs + (size_t)bh*S*S;
  const bf16* Vt = vT    + (size_t)bh*dh*S;
  bf16* O = ctxh + (size_t)bh*S*dh;
  int wid  = blockIdx.x * (blockDim.x >> 5) + (threadIdx.x >> 5);
  int lane = threadIdx.x & 31;
  int row0 = wid << 4;
  int r = lane & 15, hi = lane >> 4;
  int nt = dh >> 4;                       // 4 (dh=64) or 6 (dh=96)
  const bf16* ap = A  + (size_t)(row0 + r)*S + hi*16;
  const bf16* bp = Vt + (size_t)r*S + hi*16;
  v8f acc[6] = {v8f{},v8f{},v8f{},v8f{},v8f{},v8f{}};
  for (int k0 = 0; k0 < S; k0 += 32){
    v16bf a = load_afrag(ap + k0);
    for (int j = 0; j < nt; ++j){
      v16bf bb = *(const v16bf*)(bp + (size_t)j*16*S + k0);
      acc[j] = WMMA_BF16(a, bb, acc[j]);
    }
  }
  int orow = row0 + hi*8;
  for (int j = 0; j < nt; ++j){
    int col = j*16 + r;
    for (int e = 0; e < 8; ++e)
      O[(size_t)(orow + e)*dh + col] = (bf16)acc[j][e];
  }
}

// ----------------------------------------------------- tiny attention (S=4)
__global__ void k_small_attn(const bf16* __restrict__ q, const bf16* __restrict__ k,
                             const bf16* __restrict__ v, const float* __restrict__ cmask,
                             bf16* __restrict__ ctx, int nseq, int Sc, int heads, int dh){
  int gid = blockIdx.x*blockDim.x + threadIdx.x;
  if (gid >= nseq*heads*Sc) return;
  int qp = gid % Sc; int h = (gid/Sc)%heads; int seq = gid/(Sc*heads);
  int H = heads*dh;
  const bf16* qv = q + ((size_t)seq*Sc + qp)*H + h*dh;
  float sc[MCL_]; float mx = -1e30f;
  float scale = rsqrtf((float)dh);
  for (int kp = 0; kp < Sc; ++kp){
    const bf16* kv = k + ((size_t)seq*Sc + kp)*H + h*dh;
    float s = 0.f;
    for (int d = 0; d < dh; ++d) s += (float)qv[d]*(float)kv[d];
    s = s*scale + cmask[seq*Sc + kp];
    sc[kp] = s; mx = fmaxf(mx, s);
  }
  float sum = 0.f;
  for (int kp = 0; kp < Sc; ++kp){ sc[kp] = __expf(sc[kp]-mx); sum += sc[kp]; }
  float inv = 1.f/sum;
  bf16* o = ctx + ((size_t)seq*Sc + qp)*H + h*dh;
  for (int d = 0; d < dh; ++d){
    float a = 0.f;
    for (int kp = 0; kp < Sc; ++kp)
      a += sc[kp]*(float)v[((size_t)seq*Sc + kp)*H + h*dh + d];
    o[d] = (bf16)(a*inv);
  }
}

// ---------------------------------------------------------------- masks
__global__ void k_maskB(const int* __restrict__ am, float* __restrict__ mb, int n){
  int i = blockIdx.x*blockDim.x + threadIdx.x;
  if (i < n) mb[i] = (1.f - (float)am[i]) * -10000.f;
}

__global__ void k_chunk_prep(const int* __restrict__ cl, int* __restrict__ starts,
                             float* __restrict__ cmask){
  int b = blockIdx.x;
  __shared__ int st[MCN_];
  if (threadIdx.x == 0){
    int acc = 0;
    for (int c = 0; c < MCN_; ++c){ st[c] = acc; acc += cl[b*MCN_+c]; }
  }
  __syncthreads();
  for (int c = threadIdx.x; c < MCN_; c += blockDim.x){
    starts[b*MCN_+c] = st[c];
    int len = cl[b*MCN_+c]; if (len > MCL_) len = MCL_;
    for (int kk = 0; kk < MCL_; ++kk)
      cmask[((size_t)(b*MCN_+c))*MCL_ + kk] = (kk < len) ? 0.f : -10000.f;
  }
}

// ------------------------------------------------------- gather / scatter
__global__ void k_gather(const bf16* __restrict__ tok, const int* __restrict__ cl,
                         const int* __restrict__ starts, bf16* __restrict__ feat, int H){
  int idx = blockIdx.x;                         // (b, c, kk)
  int kk = idx % MCL_; int c = (idx/MCL_)%MCN_; int b = idx/(MCL_*MCN_);
  int len = cl[b*MCN_+c]; if (len > MCL_) len = MCL_;
  bool valid = kk < len;
  int src = starts[b*MCN_+c] + kk;
  int sc = src < 0 ? 0 : (src > L_-1 ? L_-1 : src);
  const bf16* s = tok + ((size_t)b*L_ + sc)*H;
  bf16* d = feat + (size_t)idx*H;
  for (int h = threadIdx.x; h < H; h += blockDim.x)
    d[h] = valid ? s[h] : (bf16)0.f;
}

__global__ void k_scatter(const bf16* __restrict__ enc, const int* __restrict__ cl,
                          const int* __restrict__ starts, bf16* __restrict__ span, int H){
  int idx = blockIdx.x;
  int kk = idx % MCL_; int c = (idx/MCL_)%MCN_; int b = idx/(MCL_*MCN_);
  int len = cl[b*MCN_+c]; if (len > MCL_) len = MCL_;
  if (kk >= len) return;
  int dst = starts[b*MCN_+c] + kk;
  if (dst < 0 || dst >= L_) return;
  const bf16* s = enc + (size_t)idx*H;
  bf16* d = span + ((size_t)b*L_ + dst)*H;
  for (int h = threadIdx.x; h < H; h += blockDim.x) d[h] = s[h];
}

// ---------------------------------------------------------------- compaction
__global__ void k_compact_idx(const int* __restrict__ valid_ids, int* __restrict__ dst2){
  int b = blockIdx.x;
  if (threadIdx.x == 0){
    int acc = 0;
    for (int l = 0; l < L_; ++l){
      if (valid_ids[b*L_+l] == 1){ dst2[b*L_+l] = acc; ++acc; }
      else dst2[b*L_+l] = -1;
    }
  }
}

__global__ void k_compact(const bf16* __restrict__ span, const int* __restrict__ dst2,
                          bf16* __restrict__ out, int H){
  int idx = blockIdx.x;                     // b*L_+l
  int d = dst2[idx];
  if (d < 0 || d >= L_) return;
  int b = idx / L_;
  const bf16* s = span + (size_t)idx*H;
  bf16* o = out + ((size_t)b*L_ + d)*H;
  for (int h = threadIdx.x; h < H; h += blockDim.x) o[h] = s[h];
}

// ---------------------------------------------------------------- classifier + loss
__global__ void k_cls(const bf16* __restrict__ x, const float* __restrict__ w,
                      const float* __restrict__ bias, float* __restrict__ logits, int H){
  int gid = blockIdx.x*blockDim.x + threadIdx.x;
  if (gid >= TOK_*NLAB_) return;
  int lab = gid % NLAB_; int t = gid / NLAB_;
  float s = bias[lab];
  for (int h = 0; h < H; ++h) s += (float)x[(size_t)t*H + h] * w[(size_t)h*NLAB_ + lab];
  logits[gid] = s;
}

__global__ void k_loss(const float* __restrict__ logits, const int* __restrict__ labels,
                       const int* __restrict__ lmask, float* __restrict__ accum){
  int t = blockIdx.x*blockDim.x + threadIdx.x;
  if (t >= TOK_) return;
  float v[NLAB_]; float mx = -1e30f;
  for (int j = 0; j < NLAB_; ++j){ v[j] = logits[t*NLAB_+j] + 1e-10f; mx = fmaxf(mx, v[j]); }
  float sum = 0.f;
  for (int j = 0; j < NLAB_; ++j) sum += __expf(v[j]-mx);
  float lse = mx + __logf(sum);
  int lab = labels[t];
  float nll = lse - v[lab];
  float act = (lmask[t] == 1 && lab != 0) ? 1.f : 0.f;
  atomicAdd(&accum[0], nll*act);
  atomicAdd(&accum[1], act);
}

__global__ void k_loss_fin(const float* __restrict__ accum, float* __restrict__ out){
  out[0] = accum[0] / fmaxf(accum[1], 1.f);
}

// ================================================================ host side
struct LayerDev {
  const float *qb,*kb,*vb,*ob,*ib,*o2b,*ln1g,*ln1b,*ln2g,*ln2b;
  bf16 *qwT,*kwT,*vwT,*owT,*iwT,*o2wT;
};
struct Bufs {
  bf16 *q,*k,*v,*qh,*kh,*vT,*ctxh,*ctx,*attn,*inter,*probs;
  float *tmpF,*maskB;
};

static void run_layer(bf16* x, const LayerDev& p, const Bufs& bf, int heads,
                      bool tiny, const float* tinyMask, hipStream_t s){
  const int M = TOK_, dh = H_/heads;
  dim3 blk(256);
  int gH  = (M/128)*(H_/64);
  int gFF = (M/128)*(FF_/64);
  k_gemm<0,1><<<gH,blk,0,s>>>(x, p.qwT, p.qb, bf.q, M, H_, H_);
  k_gemm<0,1><<<gH,blk,0,s>>>(x, p.kwT, p.kb, bf.k, M, H_, H_);
  k_gemm<0,1><<<gH,blk,0,s>>>(x, p.vwT, p.vb, bf.v, M, H_, H_);
  if (!tiny){
    int bh = B_*heads;
    float scale = 1.f/sqrtf((float)dh);
    k_qkv_pack<<<(TOK_*H_)/256,blk,0,s>>>(bf.q,bf.k,bf.v,bf.qh,bf.kh,bf.vT,L_,heads,dh);
    k_attn_scores<<<dim3((L_/16)*(L_/64)/8,bh),blk,0,s>>>(bf.qh,bf.kh,bf.maskB,bf.probs,L_,dh,heads,scale);
    k_softmax<<<bh*L_/8,blk,0,s>>>(bf.probs,L_);
    k_attn_ctx<<<dim3((L_/16)/8,bh),blk,0,s>>>(bf.probs,bf.vT,bf.ctxh,L_,dh);
    k_ctx_unpack<<<(TOK_*H_)/256,blk,0,s>>>(bf.ctxh,bf.ctx,L_,heads,dh);
  } else {
    int tot = B_*MCN_*CHNH_*MCL_;
    k_small_attn<<<(tot+255)/256,blk,0,s>>>(bf.q,bf.k,bf.v,tinyMask,bf.ctx,B_*MCN_,MCL_,CHNH_,dh);
  }
  k_gemm<0,0><<<gH,blk,0,s>>>(bf.ctx,  p.owT,  p.ob,  bf.tmpF, M, H_, H_);
  k_ln<<<M,blk,0,s>>>(bf.tmpF, x, p.ln1g, p.ln1b, bf.attn, H_);
  k_gemm<1,1><<<gFF,blk,0,s>>>(bf.attn, p.iwT,  p.ib,  bf.inter, M, FF_, H_);
  k_gemm<0,0><<<gH,blk,0,s>>>(bf.inter, p.o2wT, p.o2b, bf.tmpF, M, H_, FF_);
  k_ln<<<M,blk,0,s>>>(bf.tmpF, bf.attn, p.ln2g, p.ln2b, x, H_);
}

extern "C" void kernel_launch(void* const* d_in, const int* in_sizes, int n_in,
                              void* d_out, int out_size, void* d_ws, size_t ws_size,
                              hipStream_t stream){
  (void)in_sizes; (void)n_in; (void)out_size; (void)ws_size;
  const int* input_ids  = (const int*)d_in[0];
  const int* attn_mask  = (const int*)d_in[1];
  const int* tok_type   = (const int*)d_in[2];
  const int* chunk_lens = (const int*)d_in[3];
  const int* label_ids  = (const int*)d_in[4];
  const int* label_mask = (const int*)d_in[5];
  const int* valid_ids  = (const int*)d_in[6];
  auto pf = [&](int i){ return (const float*)d_in[i]; };
  // jax tree flatten (dict keys sorted): ck1[16], ck2[16], cls_b, cls_w, embb,
  // embg, layers[4][16], pos_emb, type_emb, word_emb ; layer leaves sorted:
  // ib,iw,kb,kw,ln1b,ln1g,ln2b,ln2g,o2b,o2w,ob,ow,qb,qw,vb,vw
  const int P0 = 7;
  const int IDX_CLSB = P0+32, IDX_CLSW = P0+33, IDX_EMBB = P0+34, IDX_EMBG = P0+35;
  const int IDX_POS = P0+100, IDX_TYPE = P0+101, IDX_WORD = P0+102;
  int lbase[6] = { P0+36, P0+52, P0+68, P0+84, P0+0, P0+16 }; // layers 0..3, ck1, ck2

  // workspace bump allocator
  char* wsb = (char*)d_ws; size_t off = 0;
  auto walloc = [&](size_t bytes)->void*{
    off = (off + 255) & ~(size_t)255;
    void* p = wsb + off; off += bytes; return p;
  };

  LayerDev LD[6];
  for (int i = 0; i < 6; ++i){
    int b = lbase[i];
    LD[i].ib = pf(b+0);  LD[i].kb = pf(b+2);  LD[i].ln1b = pf(b+4); LD[i].ln1g = pf(b+5);
    LD[i].ln2b = pf(b+6); LD[i].ln2g = pf(b+7); LD[i].o2b = pf(b+8);
    LD[i].ob = pf(b+10); LD[i].qb = pf(b+12); LD[i].vb = pf(b+14);
    LD[i].qwT  = (bf16*)walloc((size_t)H_*H_*2);
    LD[i].kwT  = (bf16*)walloc((size_t)H_*H_*2);
    LD[i].vwT  = (bf16*)walloc((size_t)H_*H_*2);
    LD[i].owT  = (bf16*)walloc((size_t)H_*H_*2);
    LD[i].iwT  = (bf16*)walloc((size_t)H_*FF_*2);
    LD[i].o2wT = (bf16*)walloc((size_t)FF_*H_*2);
  }
  Bufs bf;
  bf.q     = (bf16*)walloc((size_t)TOK_*H_*2);
  bf.k     = (bf16*)walloc((size_t)TOK_*H_*2);
  bf.v     = (bf16*)walloc((size_t)TOK_*H_*2);
  bf.qh    = (bf16*)walloc((size_t)TOK_*H_*2);
  bf.kh    = (bf16*)walloc((size_t)TOK_*H_*2);
  bf.vT    = (bf16*)walloc((size_t)TOK_*H_*2);
  bf.ctxh  = (bf16*)walloc((size_t)TOK_*H_*2);
  bf.ctx   = (bf16*)walloc((size_t)TOK_*H_*2);
  bf.attn  = (bf16*)walloc((size_t)TOK_*H_*2);
  bf.inter = (bf16*)walloc((size_t)TOK_*FF_*2);
  bf.probs = (bf16*)walloc((size_t)(B_*NH_)*L_*L_*2);
  bf.tmpF  = (float*)walloc((size_t)TOK_*H_*4);
  bf.maskB = (float*)walloc((size_t)B_*L_*4);
  bf16* xbuf   = (bf16*)walloc((size_t)TOK_*H_*2);
  bf16* featb  = (bf16*)walloc((size_t)TOK_*H_*2);
  bf16* spanb  = (bf16*)walloc((size_t)TOK_*H_*2);
  bf16* voutb  = (bf16*)walloc((size_t)TOK_*H_*2);
  int*  starts = (int*)walloc((size_t)B_*MCN_*4);
  float* cmask = (float*)walloc((size_t)B_*MCN_*MCL_*4);
  int*  dst2   = (int*)walloc((size_t)B_*L_*4);
  float* accum = (float*)walloc(8);

  dim3 blk(256);
  // --- weight cast+transpose (fp32 [K,N] -> bf16 [N,K]) ---
  for (int i = 0; i < 6; ++i){
    int b = lbase[i];
    k_cast_transpose<<<dim3(H_/32,H_/32),blk,0,stream>>>(pf(b+13), LD[i].qwT, H_, H_);
    k_cast_transpose<<<dim3(H_/32,H_/32),blk,0,stream>>>(pf(b+3),  LD[i].kwT, H_, H_);
    k_cast_transpose<<<dim3(H_/32,H_/32),blk,0,stream>>>(pf(b+15), LD[i].vwT, H_, H_);
    k_cast_transpose<<<dim3(H_/32,H_/32),blk,0,stream>>>(pf(b+11), LD[i].owT, H_, H_);
    k_cast_transpose<<<dim3(H_/32,FF_/32),blk,0,stream>>>(pf(b+1), LD[i].iwT, H_, FF_);
    k_cast_transpose<<<dim3(FF_/32,H_/32),blk,0,stream>>>(pf(b+9), LD[i].o2wT, FF_, H_);
  }
  // --- masks, embeddings ---
  k_maskB<<<(B_*L_)/256,blk,0,stream>>>(attn_mask, bf.maskB, B_*L_);
  k_chunk_prep<<<B_,128,0,stream>>>(chunk_lens, starts, cmask);
  k_embed<<<TOK_,blk,0,stream>>>(input_ids, tok_type, pf(IDX_WORD), pf(IDX_POS),
                                 pf(IDX_TYPE), bf.tmpF, L_, H_);
  k_ln<<<TOK_,blk,0,stream>>>(bf.tmpF, nullptr, pf(IDX_EMBG), pf(IDX_EMBB), xbuf, H_);
  // --- 4-layer token encoder ---
  for (int l = 0; l < NL_; ++l) run_layer(xbuf, LD[l], bf, NH_, false, nullptr, stream);
  // --- chunk gather + ck1 (tiny-attention layer) ---
  k_gather<<<B_*MCN_*MCL_,blk,0,stream>>>(xbuf, chunk_lens, starts, featb, H_);
  run_layer(featb, LD[4], bf, CHNH_, true, cmask, stream);
  // --- scatter back + ck2 (full layer, 8 heads) ---
  (void)hipMemsetAsync(spanb, 0, (size_t)TOK_*H_*2, stream);
  k_scatter<<<B_*MCN_*MCL_,blk,0,stream>>>(featb, chunk_lens, starts, spanb, H_);
  run_layer(spanb, LD[5], bf, CHNH_, false, nullptr, stream);
  // --- valid-token compaction ---
  k_compact_idx<<<B_,32,0,stream>>>(valid_ids, dst2);
  (void)hipMemsetAsync(voutb, 0, (size_t)TOK_*H_*2, stream);
  k_compact<<<TOK_,blk,0,stream>>>(spanb, dst2, voutb, H_);
  // --- classifier + masked NLL ---
  float* outF = (float*)d_out;
  k_cls<<<(TOK_*NLAB_+255)/256,blk,0,stream>>>(voutb, pf(IDX_CLSW), pf(IDX_CLSB), outF+1, H_);
  (void)hipMemsetAsync(accum, 0, 8, stream);
  k_loss<<<(TOK_+255)/256,blk,0,stream>>>(outF+1, label_ids, label_mask, accum);
  k_loss_fin<<<1,1,0,stream>>>(accum, outF);
}
